// SimpleRNN_73942156968551
// MI455X (gfx1250) — compile-verified, matrix operand order checked by reading, not run on hardware
//
#include <hip/hip_runtime.h>

// ---------------------------------------------------------------------------
// Gated linear-attention (GLA) style RNN, chunked for CDNA5 WMMA (gfx1250).
// T=8192, H=2048, K=V=512, O=2048, chunk C=64, NC=128 chunks.
// - All WMMA B-operands staged in LDS in [N][K] layout => A and B fragments
//   both load as two contiguous ds_load_b128 per lane (wave32 layout).
// - All global->LDS staging uses GLOBAL_LOAD_ASYNC_TO_LDS_B128 (ASYNCcnt),
//   avoiding the VGPR round-trip.
// ---------------------------------------------------------------------------

typedef __bf16 bf16_t;
typedef __attribute__((ext_vector_type(16))) __bf16 v16bf;
typedef __attribute__((ext_vector_type(8)))  float  v8f;
typedef __attribute__((ext_vector_type(8)))  unsigned short us8;
typedef __attribute__((ext_vector_type(4)))  int    v4i;

#define DEVINL static __device__ __forceinline__

static constexpr int Tn = 8192;
static constexpr int Hn = 2048;
static constexpr int KD = 512;
static constexpr int VD = 512;
static constexpr int OD = 2048;
static constexpr int Cn = 64;          // chunk length
static constexpr int NC = Tn / Cn;     // 128 chunks

// ---- helpers --------------------------------------------------------------

DEVINL unsigned short f2bf_bits(float f) {
  unsigned int u = __float_as_uint(f);
  unsigned int r = u + 0x7FFFu + ((u >> 16) & 1u);   // round-to-nearest-even
  return (unsigned short)(r >> 16);
}
DEVINL bf16_t f2bf(float f) {
  union { unsigned short u; bf16_t b; } x; x.u = f2bf_bits(f); return x.b;
}

DEVINL v8f wmma_bf16(v16bf a, v16bf b, v8f c) {
  return __builtin_amdgcn_wmma_f32_16x16x32_bf16(
      /*neg_a=*/false, a, /*neg_b=*/false, b,
      /*c_mod=*/(short)0, c, /*reuse_a=*/false, /*reuse_b=*/false);
}

// Async global->LDS copies (per-lane), tracked with ASYNCcnt.
// Builtin signatures (from toolchain diagnostics): param0 = AS1 data pointer
// (int4* for b128, int* for b32), param1 = LDS pointer, then imm offset, cpol.
template <int OFF>
DEVINL void async_g2l_b128(const void* g, void* l) {
  __builtin_amdgcn_global_load_async_to_lds_b128(
      (__attribute__((address_space(1))) v4i*)g,
      (__attribute__((address_space(3))) v4i*)l, OFF, 0);
}
template <int OFF>
DEVINL void async_g2l_b32(const void* g, void* l) {
  __builtin_amdgcn_global_load_async_to_lds_b32(
      (__attribute__((address_space(1))) int*)g,
      (__attribute__((address_space(3))) int*)l, OFF, 0);
}
DEVINL void async_wait0() { __builtin_amdgcn_s_wait_asynccnt(0); }

// Fragment from a row-major [16 rows][32 K] tile (stride lda).
// Works for A ([M][K] tile) and for B staged transposed ([N][K] tile).
DEVINL v16bf load_frag(const bf16_t* base, int lda) {
  int lane = threadIdx.x & 31;
  const bf16_t* p = base + (size_t)(lane & 15) * lda + ((lane >> 4) << 3);
  v16bf a;
#pragma unroll
  for (int i = 0; i < 8; ++i) { a[i] = p[i]; a[8 + i] = p[16 + i]; }
  return a;
}

// ---- kernel 1a: fp32 -> bf16 (same layout) --------------------------------

__global__ void cvt_f32_bf16_kernel(const float* __restrict__ src,
                                    bf16_t* __restrict__ dst, int n) {
  int i = blockIdx.x * 256 + threadIdx.x;
  if (i < n) dst[i] = f2bf(src[i]);
}

// ---- kernel 1b: fp32 [R,Nc] -> bf16 transposed [Nc,R] (tiled via LDS) -----

__global__ __launch_bounds__(256)
void transpose_cvt_kernel(const float* __restrict__ src, bf16_t* __restrict__ dst,
                          int R, int Nc) {
  __shared__ bf16_t tile[32][33];
  int r0 = blockIdx.x * 32, c0 = blockIdx.y * 32;
  int tx = threadIdx.x & 31, ty = threadIdx.x >> 5;
#pragma unroll
  for (int rr = ty; rr < 32; rr += 8)
    tile[rr][tx] = f2bf(src[(size_t)(r0 + rr) * Nc + c0 + tx]);
  __syncthreads();
#pragma unroll
  for (int rr = ty; rr < 32; rr += 8)
    dst[(size_t)(c0 + rr) * R + r0 + tx] = tile[tx][rr];
}

// ---- kernel 2: tiled BF16 GEMM, C = act(A@B + bias) -----------------------
// A [M,Kd] bf16 row-major, BT [N,Kd] bf16 row-major (i.e. B transposed).
// Output: fp32 row-major [M,N], or bf16 (row-major, or transposed [N,M]).

__global__ __launch_bounds__(256)
void gemm_bf16_kernel(const bf16_t* __restrict__ A, const bf16_t* __restrict__ BT,
                      const float* __restrict__ bias,
                      float* __restrict__ Cf, bf16_t* __restrict__ Cbf,
                      int M, int N, int Kd, int act, int transC) {
  __shared__ __align__(16) bf16_t As[128 * 32];
  __shared__ __align__(16) bf16_t Bs[128 * 32];   // [N][K] layout

  int tid  = threadIdx.x;
  int wave = tid >> 5;
  int m0 = blockIdx.x * 128;
  int n0 = blockIdx.y * 128;
  int wm = (wave & 3) * 32;     // 2 M-tiles per wave
  int wn = (wave >> 2) * 64;    // 4 N-tiles per wave

  v8f acc[2][4];
#pragma unroll
  for (int i = 0; i < 2; ++i)
#pragma unroll
    for (int j = 0; j < 4; ++j) acc[i][j] = v8f{};

  int r  = tid >> 1, ch = (tid & 1) * 16;
  for (int kk = 0; kk < Kd; kk += 32) {
    {  // stage A tile 128x32 (async global->LDS, 2x16B per lane)
      const bf16_t* srcA = A + (size_t)(m0 + r) * Kd + kk + ch;
      bf16_t* dstA = As + r * 32 + ch;
      async_g2l_b128<0>(srcA, dstA);
      async_g2l_b128<16>(srcA, dstA);
      if (kk + 32 < Kd) __builtin_prefetch(srcA + 32, 0, 1);
    }
    {  // stage BT tile 128x32 ([N][K])
      const bf16_t* srcB = BT + (size_t)(n0 + r) * Kd + kk + ch;
      bf16_t* dstB = Bs + r * 32 + ch;
      async_g2l_b128<0>(srcB, dstB);
      async_g2l_b128<16>(srcB, dstB);
    }
    async_wait0();
    __syncthreads();

    v16bf bf[4];
#pragma unroll
    for (int j = 0; j < 4; ++j) bf[j] = load_frag(Bs + (wn + j * 16) * 32, 32);
#pragma unroll
    for (int i = 0; i < 2; ++i) {
      v16bf a = load_frag(As + (wm + i * 16) * 32, 32);
#pragma unroll
      for (int j = 0; j < 4; ++j) acc[i][j] = wmma_bf16(a, bf[j], acc[i][j]);
    }
    __syncthreads();
  }

  // epilogue: bias + optional fast sigmoid
  int lane = tid & 31, n_l = lane & 15, mb = (lane >> 4) * 8;
#pragma unroll
  for (int i = 0; i < 2; ++i)
#pragma unroll
    for (int j = 0; j < 4; ++j) {
      int gm = m0 + wm + i * 16 + mb;
      int gn = n0 + wn + j * 16 + n_l;
      float bv = bias[gn];
      float vals[8];
#pragma unroll
      for (int rr = 0; rr < 8; ++rr) {
        float v = acc[i][j][rr] + bv;
        if (act == 1) v = __builtin_amdgcn_rcpf(1.0f + __expf(-v));
        vals[rr] = v;
      }
      if (Cf) {
#pragma unroll
        for (int rr = 0; rr < 8; ++rr) Cf[(size_t)(gm + rr) * N + gn] = vals[rr];
      } else if (transC) {
        us8 w;
#pragma unroll
        for (int rr = 0; rr < 8; ++rr) w[rr] = f2bf_bits(vals[rr]);
        *(us8*)(Cbf + (size_t)gn * M + gm) = w;   // [N][M], 8 contiguous
      } else {
#pragma unroll
        for (int rr = 0; rr < 8; ++rr) Cbf[(size_t)(gm + rr) * N + gn] = f2bf(vals[rr]);
      }
    }
}

// ---- kernel 3: per-chunk cumulative gate products -------------------------

__global__ void gate_cumprod_kernel(const float* __restrict__ g,
                                    float* __restrict__ Gc) {
  int idx = blockIdx.x * 256 + threadIdx.x;
  if (idx >= NC * KD) return;
  int c = idx >> 9, kk = idx & (KD - 1);
  const float* gp = g + (size_t)c * Cn * KD + kk;
  float* op = Gc + (size_t)c * Cn * KD + kk;
  float p = 1.0f;
  for (int i = 0; i < Cn; ++i) { p *= gp[(size_t)i * KD]; op[(size_t)i * KD] = p; }
}

// ---- kernel 4: pack q~ [T,K], k~ [T,K], khat^T [K,T] (bf16) ---------------

__global__ void pack_kernel(const float* __restrict__ q, const float* __restrict__ k,
                            const float* __restrict__ Gc,
                            bf16_t* __restrict__ qt, bf16_t* __restrict__ kt,
                            bf16_t* __restrict__ khT) {
  int idx = blockIdx.x * 256 + threadIdx.x;
  if (idx >= Tn * KD) return;
  int t = idx >> 9, kk = idx & (KD - 1);
  int c = t >> 6;
  float G  = fmaxf(Gc[idx], 1e-30f);
  float Ge = Gc[(size_t)(c * Cn + Cn - 1) * KD + kk];
  float kv = k[idx];
  float inv = __builtin_amdgcn_rcpf(G);
  qt[idx] = f2bf(q[idx] * G);                       // [T,K]
  kt[idx] = f2bf(kv * inv);                         // [T,K]
  khT[(size_t)kk * Tn + t] = f2bf(kv * Ge * inv);   // [K,T]
}

// ---- kernel 5: pass A — sequential state recurrence over chunks -----------
// 16 WGs, each owns a 128x128 state tile held in WMMA accumulators.
// S <- diag(Gend)*S + khat^T @ V per chunk; boundary states stored transposed
// as ST[c][v][k] bf16 (contiguous 16B stores; pass B consumes as B-operand).

__global__ __launch_bounds__(256)
void passA_state_kernel(const bf16_t* __restrict__ khT, const bf16_t* __restrict__ vT,
                        const float* __restrict__ Gc,
                        bf16_t* __restrict__ stT, float* __restrict__ finalS) {
  __shared__ __align__(16) bf16_t Kh[128 * 64];   // khat^T tile [k][t]
  __shared__ __align__(16) bf16_t Vs[128 * 64];   // V^T tile  [v][t]
  __shared__ float Ge[128];

  int tid = threadIdx.x, wave = tid >> 5, lane = tid & 31;
  int kt = blockIdx.x >> 2, vt = blockIdx.x & 3;
  int wm = (wave & 3) * 32, wn = (wave >> 2) * 64;
  int n_l = lane & 15, mb = (lane >> 4) * 8;

  v8f S[2][4];
#pragma unroll
  for (int i = 0; i < 2; ++i)
#pragma unroll
    for (int j = 0; j < 4; ++j) S[i][j] = v8f{};

  for (int c = 0; c < NC; ++c) {
    // store state *before* this chunk, transposed: stT[c][v][k]
    bf16_t* sp = stT + (size_t)c * VD * KD;
#pragma unroll
    for (int i = 0; i < 2; ++i)
#pragma unroll
      for (int j = 0; j < 4; ++j) {
        us8 w;
#pragma unroll
        for (int rr = 0; rr < 8; ++rr) w[rr] = f2bf_bits(S[i][j][rr]);
        *(us8*)(sp + (size_t)(vt * 128 + wn + j * 16 + n_l) * KD +
                kt * 128 + wm + i * 16 + mb) = w;
      }

    // stage khat^T [128k x 64t] and V^T [128v x 64t] (async, contiguous 16B)
    for (int i = tid; i < 1024; i += 256) {
      int row = i >> 3, cc = (i & 7) * 8;
      async_g2l_b128<0>(khT + (size_t)(kt * 128 + row) * Tn + c * Cn + cc, Kh + i * 8);
      async_g2l_b128<0>(vT + (size_t)(vt * 128 + row) * Tn + c * Cn + cc, Vs + i * 8);
    }
    if (tid < 128)
      async_g2l_b32<0>(Gc + (size_t)(c * Cn + Cn - 1) * KD + kt * 128 + tid, Ge + tid);
    async_wait0();
    __syncthreads();

    // decay state, then accumulate k_hat^T @ V via WMMA
#pragma unroll
    for (int i = 0; i < 2; ++i)
#pragma unroll
      for (int j = 0; j < 4; ++j)
#pragma unroll
        for (int rr = 0; rr < 8; ++rr)
          S[i][j][rr] *= Ge[wm + i * 16 + mb + rr];

#pragma unroll
    for (int ks = 0; ks < 64; ks += 32) {
      v16bf bf[4];
#pragma unroll
      for (int j = 0; j < 4; ++j) bf[j] = load_frag(Vs + (wn + j * 16) * 64 + ks, 64);
#pragma unroll
      for (int i = 0; i < 2; ++i) {
        v16bf a = load_frag(Kh + (size_t)(wm + i * 16) * 64 + ks, 64);
#pragma unroll
        for (int j = 0; j < 4; ++j) S[i][j] = wmma_bf16(a, bf[j], S[i][j]);
      }
    }
    __syncthreads();
  }

  // final state (fp32, [K][V] row-major) straight into d_out tail
#pragma unroll
  for (int i = 0; i < 2; ++i)
#pragma unroll
    for (int j = 0; j < 4; ++j)
#pragma unroll
      for (int rr = 0; rr < 8; ++rr)
        finalS[(size_t)(kt * 128 + wm + i * 16 + mb + rr) * VD +
               vt * 128 + wn + j * 16 + n_l] = S[i][j][rr];
}

// ---- kernel 6: pass B — per-chunk output -----------------------------------
// out = q~ @ S_c + tril(q~ @ k~^T) @ V ; one WG per (chunk, 128-wide V tile).
// Intra-chunk matrix computed as its transpose (k~ @ q~^T) so both operands
// stage contiguously and the masked result stores contiguously.

__global__ __launch_bounds__(256)
void passB_output_kernel(const bf16_t* __restrict__ qt, const bf16_t* __restrict__ kt,
                         const bf16_t* __restrict__ vT, const bf16_t* __restrict__ stT,
                         bf16_t* __restrict__ outs) {
  __shared__ __align__(16) bf16_t As1[64 * 32];
  __shared__ __align__(16) bf16_t Bs[128 * 32];   // [v][k] from stT
  __shared__ __align__(16) bf16_t Bs2[64 * 32];   // [t][k] from qt
  __shared__ __align__(16) bf16_t A2[64 * 64];    // masked Aintra [t][t']
  __shared__ __align__(16) bf16_t Vs[128 * 64];   // [v][t'] from vT

  int c = blockIdx.x, vt = blockIdx.y;
  int tid = threadIdx.x, wave = tid >> 5, lane = tid & 31;
  int t0 = c * Cn;
  int wm = (wave & 3) * 16;     // 1 M-tile per wave (t rows)
  int wn = (wave >> 2) * 64;    // 4 N-tiles per wave (v cols)
  int n_l = lane & 15, mb = (lane >> 4) * 8;

  v8f acc[4];
#pragma unroll
  for (int j = 0; j < 4; ++j) acc[j] = v8f{};

  // --- step 1: O1 = q~ @ S_c[:, vtile]  (M=64 t, N=128 v, K=512 k)
  for (int ks = 0; ks < KD; ks += 32) {
    {
      int row = tid >> 2, cc = (tid & 3) * 8;
      async_g2l_b128<0>(qt + (size_t)(t0 + row) * KD + ks + cc, As1 + tid * 8);
    }
    {
      int row = tid >> 1, ch = (tid & 1) * 16;
      const bf16_t* src = stT + (size_t)c * VD * KD +
                          (size_t)(vt * 128 + row) * KD + ks + ch;
      bf16_t* dst = Bs + row * 32 + ch;
      async_g2l_b128<0>(src, dst);
      async_g2l_b128<16>(src, dst);
    }
    async_wait0();
    __syncthreads();
    v16bf a = load_frag(As1 + wm * 32, 32);
#pragma unroll
    for (int j = 0; j < 4; ++j) {
      v16bf b = load_frag(Bs + (wn + j * 16) * 32, 32);
      acc[j] = wmma_bf16(a, b, acc[j]);
    }
    __syncthreads();
  }

  // --- step 2: Aintra^T = k~ @ q~^T  (M=64 t', N=64 t, K=512 k), masked
  v8f acc2[2];
#pragma unroll
  for (int j = 0; j < 2; ++j) acc2[j] = v8f{};
  int wm2 = (wave & 3) * 16;    // t' rows
  int wn2 = (wave >> 2) * 32;   // t cols, 2 tiles

  for (int ks = 0; ks < KD; ks += 32) {
    {
      int row = tid >> 2, cc = (tid & 3) * 8;
      async_g2l_b128<0>(kt + (size_t)(t0 + row) * KD + ks + cc, As1 + tid * 8);
      async_g2l_b128<0>(qt + (size_t)(t0 + row) * KD + ks + cc, Bs2 + tid * 8);
    }
    async_wait0();
    __syncthreads();
    v16bf a = load_frag(As1 + wm2 * 32, 32);
#pragma unroll
    for (int j = 0; j < 2; ++j) {
      v16bf b = load_frag(Bs2 + (wn2 + j * 16) * 32, 32);
      acc2[j] = wmma_bf16(a, b, acc2[j]);
    }
    __syncthreads();
  }

  // causal mask (keep t >= t'); acc2 holds Aintra^T so transpose-store into
  // row-major A2[t][t'] => contiguous 16B LDS stores per lane.
#pragma unroll
  for (int j = 0; j < 2; ++j) {
    int col_t = wn2 + j * 16 + n_l;   // t
    us8 w;
#pragma unroll
    for (int rr = 0; rr < 8; ++rr) {
      int row_tp = wm2 + mb + rr;     // t'
      w[rr] = f2bf_bits((col_t >= row_tp) ? acc2[j][rr] : 0.0f);
    }
    *(us8*)(A2 + col_t * 64 + wm2 + mb) = w;
  }
  // stage V^T tile [128v x 64t'] (async)
  for (int i = tid; i < 1024; i += 256) {
    int row = i >> 3, cc = (i & 7) * 8;
    async_g2l_b128<0>(vT + (size_t)(vt * 128 + row) * Tn + t0 + cc, Vs + i * 8);
  }
  async_wait0();
  __syncthreads();

  // --- step 3: O += tril(Aintra) @ V  (M=64 t, N=128 v, K=64 t')
#pragma unroll
  for (int ks = 0; ks < 64; ks += 32) {
    v16bf a = load_frag(A2 + wm * 64 + ks, 64);
#pragma unroll
    for (int j = 0; j < 4; ++j) {
      v16bf b = load_frag(Vs + (wn + j * 16) * 64 + ks, 64);
      acc[j] = wmma_bf16(a, b, acc[j]);
    }
  }

  // write outs (bf16 [T,V], feeds final WMMA GEMM)
#pragma unroll
  for (int j = 0; j < 4; ++j)
#pragma unroll
    for (int rr = 0; rr < 8; ++rr)
      outs[(size_t)(t0 + wm + mb + rr) * VD + vt * 128 + wn + j * 16 + n_l] =
          f2bf(acc[j][rr]);
}

// ---------------------------------------------------------------------------

extern "C" void kernel_launch(void* const* d_in, const int* in_sizes, int n_in,
                              void* d_out, int out_size, void* d_ws, size_t ws_size,
                              hipStream_t stream) {
  const float* hidden = (const float*)d_in[0];
  const float* Wq = (const float*)d_in[1];  const float* bq = (const float*)d_in[2];
  const float* Wk = (const float*)d_in[3];  const float* bk = (const float*)d_in[4];
  const float* Wv = (const float*)d_in[5];  const float* bv = (const float*)d_in[6];
  const float* Wg = (const float*)d_in[7];  const float* bg = (const float*)d_in[8];
  const float* Wo = (const float*)d_in[9];  const float* bo = (const float*)d_in[10];

  float* out_f = (float*)d_out;                    // [T,O] output
  float* final_state = out_f + (size_t)Tn * OD;    // [K,V] final state

  size_t off = 0;
  auto wsalloc = [&](size_t bytes) -> void* {
    void* p = (char*)d_ws + off;
    off = (off + bytes + 255) & ~(size_t)255;
    return p;
  };
  bf16_t* hid_bf = (bf16_t*)wsalloc((size_t)Tn * Hn * 2);
  bf16_t* wqT_bf = (bf16_t*)wsalloc((size_t)Hn * KD * 2);   // [K,H]
  bf16_t* wkT_bf = (bf16_t*)wsalloc((size_t)Hn * KD * 2);
  bf16_t* wvT_bf = (bf16_t*)wsalloc((size_t)Hn * VD * 2);
  bf16_t* wgT_bf = (bf16_t*)wsalloc((size_t)Hn * KD * 2);
  bf16_t* woT_bf = (bf16_t*)wsalloc((size_t)VD * OD * 2);   // [O,V]
  float*  qf     = (float*)wsalloc((size_t)Tn * KD * 4);
  float*  kf     = (float*)wsalloc((size_t)Tn * KD * 4);
  float*  gf     = (float*)wsalloc((size_t)Tn * KD * 4);
  float*  Gcum   = (float*)wsalloc((size_t)Tn * KD * 4);
  bf16_t* vT_bf  = (bf16_t*)wsalloc((size_t)Tn * VD * 2);   // [V,T]
  bf16_t* qt_bf  = (bf16_t*)wsalloc((size_t)Tn * KD * 2);   // [T,K]
  bf16_t* kt_bf  = (bf16_t*)wsalloc((size_t)Tn * KD * 2);   // [T,K]
  bf16_t* khT_bf = (bf16_t*)wsalloc((size_t)KD * Tn * 2);   // [K,T]
  bf16_t* st_bf  = (bf16_t*)wsalloc((size_t)NC * KD * VD * 2); // [c][V][K]
  bf16_t* o_bf   = (bf16_t*)wsalloc((size_t)Tn * VD * 2);   // [T,V]

  // 1. convert/transpose operands to bf16
  cvt_f32_bf16_kernel<<<(Tn * Hn + 255) / 256, 256, 0, stream>>>(hidden, hid_bf, Tn * Hn);
  transpose_cvt_kernel<<<dim3(Hn / 32, KD / 32), 256, 0, stream>>>(Wq, wqT_bf, Hn, KD);
  transpose_cvt_kernel<<<dim3(Hn / 32, KD / 32), 256, 0, stream>>>(Wk, wkT_bf, Hn, KD);
  transpose_cvt_kernel<<<dim3(Hn / 32, VD / 32), 256, 0, stream>>>(Wv, wvT_bf, Hn, VD);
  transpose_cvt_kernel<<<dim3(Hn / 32, KD / 32), 256, 0, stream>>>(Wg, wgT_bf, Hn, KD);
  transpose_cvt_kernel<<<dim3(VD / 32, OD / 32), 256, 0, stream>>>(Wo, woT_bf, VD, OD);

  // 2. projections (WMMA GEMMs): q, k=sigmoid, g=sigmoid, v (bf16^T direct)
  dim3 gP(Tn / 128, KD / 128);
  gemm_bf16_kernel<<<gP, 256, 0, stream>>>(hid_bf, wqT_bf, bq, qf, nullptr, Tn, KD, Hn, 0, 0);
  gemm_bf16_kernel<<<gP, 256, 0, stream>>>(hid_bf, wkT_bf, bk, kf, nullptr, Tn, KD, Hn, 1, 0);
  gemm_bf16_kernel<<<gP, 256, 0, stream>>>(hid_bf, wgT_bf, bg, gf, nullptr, Tn, KD, Hn, 1, 0);
  gemm_bf16_kernel<<<gP, 256, 0, stream>>>(hid_bf, wvT_bf, bv, nullptr, vT_bf, Tn, VD, Hn, 0, 1);

  // 3. cumulative gate products + operand packing
  gate_cumprod_kernel<<<(NC * KD + 255) / 256, 256, 0, stream>>>(gf, Gcum);
  pack_kernel<<<(Tn * KD + 255) / 256, 256, 0, stream>>>(qf, kf, Gcum, qt_bf, kt_bf, khT_bf);

  // 4. pass A: sequential chunk recurrence, state tiles in WMMA accumulators
  passA_state_kernel<<<16, 256, 0, stream>>>(khT_bf, vT_bf, Gcum, st_bf, final_state);

  // 5. pass B: all chunks in parallel (inter-chunk + masked intra-chunk)
  passB_output_kernel<<<dim3(NC, VD / 128), 256, 0, stream>>>(qt_bf, kt_bf, vT_bf, st_bf, o_bf);

  // 6. output projection: outs @ Wo + bo -> d_out (fp32)
  dim3 gO(Tn / 128, OD / 128);
  gemm_bf16_kernel<<<gO, 256, 0, stream>>>(o_bf, woT_bf, bo, out_f, nullptr, Tn, OD, KD, 0, 0);

  (void)in_sizes; (void)n_in; (void)out_size; (void)ws_size;
}
